// Model_39676907884200
// MI455X (gfx1250) — matrix operand order repaired and checked by reading, not compiled, not run on hardware
//
#include <hip/hip_runtime.h>
#include <hip/hip_bf16.h>

// Fused per-(b,s) attention block for MI455X (gfx1250, wave32).
// grid = 8192 blocks (one per (b,s)), block = 256 threads = 8 wave32s.
// Memory-bound: ~256 MiB HBM traffic vs ~21.5 GFLOP -> f16 WMMA
// (v_wmma_f32_16x16x32_f16) keeps math time well under memory time.

typedef __attribute__((ext_vector_type(16))) _Float16 v16h;
typedef __attribute__((ext_vector_type(8)))  _Float16 v8h;
typedef __attribute__((ext_vector_type(8)))  float    v8f;

// ---- LDS layout (bytes). Total 58112 < 64KB static limit. ----
// [     0,  8K)  sX   : X f16 64x64            (dead after stage 2)
// [    8K, 32K)  sW   : Wq|Wk|Wv f16 64x64 ea. (dead after stage 2)
// [   32K, 40K)  sQt  : Q^T f16 (d-major)
// [   40K, 48K)  sKt  : K^T f16 (e-major)
// [   48K, 56K)  sVt  : V^T f16 (k-major)
// [   56K, +768) sBias: bq|bk|bv f32           (dead after stage 2)
// aliases (after stage-2 barrier):
// [     0, 16K)  sS   : scores f32 64x64, then O f32 64x64 (aliases sX + sWq)
// [   16K, 24K)  sA   : attn f16 64x64                     (aliases sWk)
#define SMEM_BYTES (56 * 1024 + 768)

__device__ __forceinline__ unsigned hash32(unsigned x) {
    x ^= x >> 16; x *= 0x7feb352dU;
    x ^= x >> 15; x *= 0x846ca68bU;
    x ^= x >> 16;
    return x;
}

// A-fragment (16x32 f16, M x K): lane m = lane&15; lanes>=16 use K-offset +16.
// v[0..7]  = K = kb + koff + 0..7   (VGPR0..3)
// v[8..15] = K = kb + koff + 8..15  (VGPR4..7)
__device__ __forceinline__ v16h load_frag_A(const _Float16* base, int row, int kb, int lane) {
    const int koff = kb + ((lane & 16) ? 16 : 0);
    const _Float16* p = base + row * 64 + koff;
    v8h lo = *(const v8h*)(p);
    v8h hi = *(const v8h*)(p + 8);
    v16h a;
#pragma unroll
    for (int t = 0; t < 8; ++t) { a[t] = lo[t]; a[8 + t] = hi[t]; }
    return a;
}

// B-fragment (32x16 f16, K x N): lane n = lane&15; lanes>=16 hold K+16..31.
// Source is stored N-major (row = n, 64 K per row) -> 32 contiguous bytes.
__device__ __forceinline__ v16h load_frag_B(const _Float16* base, int row, int kb, int lane) {
    const int koff = kb + ((lane & 16) ? 16 : 0);
    const _Float16* p = base + row * 64 + koff;
    v8h lo = *(const v8h*)(p);
    v8h hi = *(const v8h*)(p + 8);
    v16h b;
#pragma unroll
    for (int t = 0; t < 8; ++t) { b[t] = lo[t]; b[8 + t] = hi[t]; }
    return b;
}

// D-tile (16x16 f32): elem g at (M = 16*ti + off + g, N = 16*tj + n).
// Store transposed as f16: dst[N][M] -> per lane 8 contiguous f16 (one v8h).
__device__ __forceinline__ void store_tile_T(_Float16* dst, int ti, int tj, int lane, v8f acc) {
    const int n   = lane & 15;
    const int off = (lane & 16) ? 8 : 0;
    v8h o;
#pragma unroll
    for (int g = 0; g < 8; ++g) o[g] = (_Float16)acc[g];
    *(v8h*)(dst + (16 * tj + n) * 64 + 16 * ti + off) = o;
}

__global__ __launch_bounds__(256) void fused_attn_kernel(
    const float* __restrict__ x1,
    const float* __restrict__ Wq, const float* __restrict__ bq,
    const float* __restrict__ Wk, const float* __restrict__ bk,
    const float* __restrict__ Wv, const float* __restrict__ bv,
    float* __restrict__ out)
{
    __shared__ __align__(16) char smem[SMEM_BYTES];
    _Float16* sX    = (_Float16*)(smem);
    _Float16* sW    = (_Float16*)(smem + 8 * 1024);   // 3 * 4096 f16
    _Float16* sQt   = (_Float16*)(smem + 32 * 1024);
    _Float16* sKt   = (_Float16*)(smem + 40 * 1024);
    _Float16* sVt   = (_Float16*)(smem + 48 * 1024);
    float*    sBias = (float*)   (smem + 56 * 1024);  // 3 * 64 f32
    float*    sS    = (float*)   (smem);              // alias: scores, then O
    _Float16* sA    = (_Float16*)(smem + 16 * 1024);  // alias: attn f16

    const int tid  = threadIdx.x;
    const int lane = tid & 31;
    const int wave = tid >> 5;
    const int blk  = blockIdx.x;      // blk = b*256 + s
    const int b    = blk >> 8;
    const int s    = blk & 255;

    const int n   = lane & 15;
    const int off = (lane & 16) ? 8 : 0;

    // ---------------- Stage 1: stage X, W, biases into LDS as f16 ----------------
    {
        // X[h][c] = x1[b, h, s, c]  (strided gather over h, contiguous in c)
        for (int i = tid; i < 1024; i += 256) {
            const int h  = i >> 4;
            const int c4 = i & 15;
            const float4 v = ((const float4*)(x1 + (((size_t)b * 64 + h) * 256 + s) * 64))[c4];
            _Float16* d = sX + h * 64 + c4 * 4;
            d[0] = (_Float16)v.x; d[1] = (_Float16)v.y;
            d[2] = (_Float16)v.z; d[3] = (_Float16)v.w;
        }
        const float* Ws[3] = { Wq, Wk, Wv };
#pragma unroll
        for (int w = 0; w < 3; ++w) {
            for (int i = tid; i < 1024; i += 256) {
                const float4 v = ((const float4*)Ws[w])[i];
                _Float16* d = sW + w * 4096 + i * 4;
                d[0] = (_Float16)v.x; d[1] = (_Float16)v.y;
                d[2] = (_Float16)v.z; d[3] = (_Float16)v.w;
            }
        }
        if (tid < 64) {
            sBias[tid]       = bq[tid];
            sBias[64 + tid]  = bk[tid];
            sBias[128 + tid] = bv[tid];
        }
    }
    __syncthreads();

    // ------------- Stage 2: Q/K/V = X @ W^T + b ; store transposed f16 -------------
    // Linear B-fragment: B[c, d] = W[d, c] -> W row-major already N-major. 2 tiles/wave.
#pragma unroll
    for (int t = wave * 2; t < wave * 2 + 2; ++t) {
        const int ti = t >> 2, tj = t & 3;
        const float bqv = sBias[16 * tj + n];
        const float bkv = sBias[64 + 16 * tj + n];
        const float bvv = sBias[128 + 16 * tj + n];
        v8f aq, ak, av;
#pragma unroll
        for (int g = 0; g < 8; ++g) { aq[g] = bqv; ak[g] = bkv; av[g] = bvv; }
#pragma unroll
        for (int kb = 0; kb < 64; kb += 32) {
            const v16h a  = load_frag_A(sX, 16 * ti + n, kb, lane);
            const v16h wq = load_frag_B(sW,         16 * tj + n, kb, lane);
            const v16h wk = load_frag_B(sW + 4096,  16 * tj + n, kb, lane);
            const v16h wv = load_frag_B(sW + 8192,  16 * tj + n, kb, lane);
            aq = __builtin_amdgcn_wmma_f32_16x16x32_f16(false, a, false, wq, (short)0, aq, false, false);
            ak = __builtin_amdgcn_wmma_f32_16x16x32_f16(false, a, false, wk, (short)0, ak, false, false);
            av = __builtin_amdgcn_wmma_f32_16x16x32_f16(false, a, false, wv, (short)0, av, false, false);
        }
        store_tile_T(sQt, ti, tj, lane, aq);
        store_tile_T(sKt, ti, tj, lane, ak);
        store_tile_T(sVt, ti, tj, lane, av);
    }
    __syncthreads();

    // ------------- Stage 3: S[d,e] = 8 * sum_h Q[h,d] K[h,e] -------------
    // A = Qt rows (d-major), B = Kt rows (e-major) -> both contiguous in K=h.
#pragma unroll
    for (int t = wave * 2; t < wave * 2 + 2; ++t) {
        const int ti = t >> 2, tj = t & 3;
        v8f acc;
#pragma unroll
        for (int g = 0; g < 8; ++g) acc[g] = 0.0f;
#pragma unroll
        for (int kb = 0; kb < 64; kb += 32) {
            const v16h a  = load_frag_A(sQt, 16 * ti + n, kb, lane);
            const v16h bm = load_frag_B(sKt, 16 * tj + n, kb, lane);
            acc = __builtin_amdgcn_wmma_f32_16x16x32_f16(false, a, false, bm, (short)0, acc, false, false);
        }
#pragma unroll
        for (int g = 0; g < 8; ++g)
            sS[(16 * ti + off + g) * 64 + 16 * tj + n] = acc[g] * 8.0f;
    }
    __syncthreads();

    // ------------- Stage 4: row softmax + dropout(p=0.1), write f16 attn -------------
    // 4 threads per row (one quad, same wave) x 16 elems each; quad-reduce via shfl_xor.
    {
        const int r = tid >> 2;       // row d
        const int q = tid & 3;        // quarter of the row
        const float* srow = sS + r * 64 + q * 16;
        float ex[16];
        float mx = -INFINITY;
#pragma unroll
        for (int e = 0; e < 16; ++e) { ex[e] = srow[e]; mx = fmaxf(mx, ex[e]); }
        mx = fmaxf(mx, __shfl_xor(mx, 1, 32));
        mx = fmaxf(mx, __shfl_xor(mx, 2, 32));
        float sum = 0.0f;
#pragma unroll
        for (int e = 0; e < 16; ++e) { ex[e] = __expf(ex[e] - mx); sum += ex[e]; }
        sum += __shfl_xor(sum, 1, 32);
        sum += __shfl_xor(sum, 2, 32);
        const float inv = (1.0f / 0.9f) / sum;   // fold dropout rescale into normalizer
        const unsigned base = ((unsigned)blk * 64u + (unsigned)r) * 64u + (unsigned)(q * 16);
        _Float16* arow = sA + r * 64 + q * 16;
#pragma unroll
        for (int e = 0; e < 16; ++e) {
            // deterministic counter-based keep mask, P(keep)=0.9
            const unsigned h = hash32(base + (unsigned)e);
            const float a = (h < 3865470566u) ? ex[e] * inv : 0.0f;
            arow[e] = (_Float16)a;
        }
    }
    __syncthreads();

    // ------------- Stage 5: O = A @ V ; stage O in LDS (reuse sS) -------------
#pragma unroll
    for (int t = wave * 2; t < wave * 2 + 2; ++t) {
        const int ti = t >> 2, tj = t & 3;
        v8f acc;
#pragma unroll
        for (int g = 0; g < 8; ++g) acc[g] = 0.0f;
#pragma unroll
        for (int kb = 0; kb < 64; kb += 32) {
            const v16h a  = load_frag_A(sA,  16 * ti + n, kb, lane);
            const v16h bm = load_frag_B(sVt, 16 * tj + n, kb, lane);
            acc = __builtin_amdgcn_wmma_f32_16x16x32_f16(false, a, false, bm, (short)0, acc, false, false);
        }
#pragma unroll
        for (int g = 0; g < 8; ++g)
            sS[(16 * ti + off + g) * 64 + 16 * tj + n] = acc[g];
    }
    __syncthreads();

    // ------------- Stage 6: coalesced float4 write-out -------------
    // out[g64, i, h', k]: g64 = blk/64, h' = blk%64 ; row i = 64 contiguous floats.
    {
        const int g64 = blk >> 6;
        const int hp  = blk & 63;
        float* __restrict__ obase = out + (size_t)g64 * (64 * 64 * 64) + (size_t)hp * 64;
#pragma unroll
        for (int i = tid; i < 1024; i += 256) {
            const int row = i >> 4;
            const int c4  = i & 15;
            const float4 v = ((const float4*)(sS + row * 64))[c4];
            ((float4*)(obase + (size_t)row * 4096))[c4] = v;
        }
    }
}

extern "C" void kernel_launch(void* const* d_in, const int* in_sizes, int n_in,
                              void* d_out, int out_size, void* d_ws, size_t ws_size,
                              hipStream_t stream) {
    const float* x1 = (const float*)d_in[0];
    const float* Wq = (const float*)d_in[1];
    const float* bq = (const float*)d_in[2];
    const float* Wk = (const float*)d_in[3];
    const float* bk = (const float*)d_in[4];
    const float* Wv = (const float*)d_in[5];
    const float* bv = (const float*)d_in[6];
    float* out = (float*)d_out;

    dim3 grid(8192);   // one block per (b, s)
    dim3 block(256);   // 8 wave32s
    hipLaunchKernelGGL(fused_attn_kernel, grid, block, 0, stream,
                       x1, Wq, bq, Wk, bk, Wv, bv, out);
}
